// RelativePartialLearnableMultiHeadAttention_87308095193117
// MI455X (gfx1250) — matrix operand order repaired and checked by reading, not compiled, not run on hardware
//
#include <hip/hip_runtime.h>

// ---------------------------------------------------------------------------
// Transformer-XL relative multi-head attention for MI455X (gfx1250, wave32).
// GEMMs: v_wmma_f32_16x16x32_bf16 with double-buffered async-to-LDS staging.
// Flash-style attention with online softmax. fp32 in/out; bf16 internal.
// ---------------------------------------------------------------------------

#define D_MODEL 1024
#define N_HEAD  16
#define D_HEAD  64
#define Q_LEN   2048
#define MEM_LEN 2048
#define K_LEN   (MEM_LEN + Q_LEN)   // 4096

typedef __attribute__((ext_vector_type(16))) __bf16 v16bf;
typedef __attribute__((ext_vector_type(8)))  __bf16 v8bf;
typedef __attribute__((ext_vector_type(8)))  float  v8f;
typedef __attribute__((ext_vector_type(4)))  int    v4i;

// ---------------------------------------------------------------------------
// CDNA5 async-to-LDS copy (GLOBAL_LOAD_ASYNC_TO_LDS_B128, ASYNCcnt-tracked)
// Probe-derived signature: (v4i addrspace(1)*, v4i addrspace(3)*, imm, imm)
// ---------------------------------------------------------------------------

#if defined(__has_builtin)
#if __has_builtin(__builtin_amdgcn_global_load_async_to_lds_b128)
#define HAVE_ASYNC_LDS 1
#endif
#endif

typedef __attribute__((address_space(1))) v4i* as1_v4i;
typedef __attribute__((address_space(3))) v4i* as3_v4i;

// Copy 16 bytes global -> LDS. Async path lands via ASYNCcnt; fallback is a
// synchronous register round-trip (made visible by the same __syncthreads()).
__device__ __forceinline__ void copy16_g2l(const __bf16* g, __bf16* l) {
#if defined(HAVE_ASYNC_LDS)
  // Flat shared address truncated to 32 bits == LDS offset (ISA 10.2 aperture).
  __builtin_amdgcn_global_load_async_to_lds_b128(
      (as1_v4i)(uintptr_t)g,
      (as3_v4i)(uintptr_t)(unsigned)(uintptr_t)l,
      /*offset=*/0, /*cpol=*/0);
#else
  *(int4*)l = *(const int4*)g;
#endif
}

// Wait until at most `pending` stages (8 async ops each) remain in flight.
__device__ __forceinline__ void wait_async_stage(bool one_pending) {
#if defined(HAVE_ASYNC_LDS)
#if __has_builtin(__builtin_amdgcn_s_wait_asynccnt)
  if (one_pending) __builtin_amdgcn_s_wait_asynccnt(8);
  else             __builtin_amdgcn_s_wait_asynccnt(0);
#else
  if (one_pending) asm volatile("s_wait_asynccnt 0x8" ::: "memory");
  else             asm volatile("s_wait_asynccnt 0x0" ::: "memory");
#endif
#else
  (void)one_pending;
#endif
}

// ---------------------------------------------------------------------------
// WMMA helpers (layouts per CDNA5 ISA 7.12.2)
// ---------------------------------------------------------------------------

__device__ __forceinline__ v8f wmma_bf16(v16bf a, v16bf b, v8f c) {
  return __builtin_amdgcn_wmma_f32_16x16x32_bf16(
      /*neg_a=*/false, a, /*neg_b=*/false, b,
      /*c_mod=*/(short)0, c, /*reuse_a=*/false, /*reuse_b=*/false);
}

// A-matrix 16x32 bf16 fragment, row-major source with leading dim ld.
// Lane L (r=L&15,g=L>>4): VGPR0..3 hold K=g*8+0..7, VGPR4..7 hold K=16+g*8+0..7.
__device__ __forceinline__ v16bf load_a_frag(const __bf16* base, int ld, int k0) {
  const int lane = threadIdx.x & 31;
  const int r = lane & 15, g = lane >> 4;
  const __bf16* p = base + r * ld + k0 + g * 8;
  v8bf lo = *(const v8bf*)(p);
  v8bf hi = *(const v8bf*)(p + 16);
  return __builtin_shufflevector(lo, hi, 0,1,2,3,4,5,6,7,8,9,10,11,12,13,14,15);
}

// B-matrix 32x16 bf16 fragment from an N-major array srcT[n][k], leading dim ldt.
// Lane L (n=L&15,g=L>>4) holds K = g*8+0..7 and K = 16+g*8+0..7 of column n.
__device__ __forceinline__ v16bf load_b_frag(const __bf16* baseT, int ldt, int k0) {
  const int lane = threadIdx.x & 31;
  const int n = lane & 15, g = lane >> 4;
  const __bf16* p = baseT + n * ldt + k0 + g * 8;
  v8bf lo = *(const v8bf*)(p);
  v8bf hi = *(const v8bf*)(p + 16);
  return __builtin_shufflevector(lo, hi, 0,1,2,3,4,5,6,7,8,9,10,11,12,13,14,15);
}

// C/D 16x16 f32: element (M = v + 8*(lane>=16), N = lane&15) in VGPR v.
__device__ __forceinline__ void store_c_frag(float* base, int ldc, v8f c) {
  const int lane = threadIdx.x & 31;
  const int n = lane & 15, g = lane >> 4;
#pragma unroll
  for (int v = 0; v < 8; ++v) base[(v + 8 * g) * (long)ldc + n] = c[v];
}

__device__ __forceinline__ v8f zero8() { v8f z = {}; return z; }

// ---------------------------------------------------------------------------
// Elementwise convert / transpose-convert
// ---------------------------------------------------------------------------

__global__ void cvt_f32_bf16(const float* __restrict__ src, __bf16* __restrict__ dst, int n) {
  int i = blockIdx.x * blockDim.x + threadIdx.x;
  if (i < n) dst[i] = (__bf16)src[i];
}

// src: Rows x Cols (f32, row-major)  ->  dst: Cols x Rows (bf16, row-major)
__global__ __launch_bounds__(256) void transpose_cvt(const float* __restrict__ src,
                                                     __bf16* __restrict__ dst,
                                                     int Rows, int Cols) {
  __shared__ float t[32][33];
  const int c0 = blockIdx.x * 32, r0 = blockIdx.y * 32;
  const int tx = threadIdx.x, ty = threadIdx.y;
  for (int i = ty; i < 32; i += 8)
    t[i][tx] = src[(long)(r0 + i) * Cols + c0 + tx];
  __syncthreads();
  for (int i = ty; i < 32; i += 8)
    dst[(long)(c0 + i) * Rows + r0 + tx] = (__bf16)t[tx][i];
}

// ---------------------------------------------------------------------------
// bf16 GEMM: C[M,N] (f32) = A[M,K] (bf16 row-major) * BT[N,K] (bf16 N-major).
// Block tile 128x128, K-stage 64, double-buffered LDS fed by async B128 copies.
// 256 threads = 8 waves (2 M x 4 N); per-wave tile 64x32 (4x2 fragments).
// LDS rows padded to 72 bf16 (36 dwords): 16 lanes * 36 dwords -> 16 distinct
// bank groups, conflict-free ds_load_b128 fragment reads.
// ---------------------------------------------------------------------------

#define GEMM_KSTAGE 64
#define LDS_PAD     72

__device__ __forceinline__ void stage_load(const __bf16* __restrict__ A,
                                           const __bf16* __restrict__ BT,
                                           int K, int m0blk, int n0blk, int ks,
                                           __bf16 (*tA)[LDS_PAD], __bf16 (*tB)[LDS_PAD]) {
  const int tid = threadIdx.x;
  const long kbase = (long)ks * GEMM_KSTAGE;
#pragma unroll
  for (int i = 0; i < 4; ++i) {           // 128 rows x 64 cols = 1024 x 16B chunks
    const int chunk = tid + i * 256;
    const int row = chunk >> 3;
    const int col = (chunk & 7) * 8;
    copy16_g2l(A + (long)(m0blk + row) * K + kbase + col, &tA[row][col]);
  }
#pragma unroll
  for (int i = 0; i < 4; ++i) {
    const int chunk = tid + i * 256;
    const int row = chunk >> 3;
    const int col = (chunk & 7) * 8;
    copy16_g2l(BT + (long)(n0blk + row) * K + kbase + col, &tB[row][col]);
  }
}

__global__ __launch_bounds__(256) void gemm_bf16(const __bf16* __restrict__ A,
                                                 const __bf16* __restrict__ BT,
                                                 float* __restrict__ C,
                                                 int M, int N, int K) {
  __shared__ __align__(16) __bf16 tileA[2][128][LDS_PAD];   // 2 x 18 KB
  __shared__ __align__(16) __bf16 tileB[2][128][LDS_PAD];   // 2 x 18 KB

  const int wave = threadIdx.x >> 5;
  const int wm = wave >> 2, wn = wave & 3;
  const int m0blk = blockIdx.y * 128;
  const int n0blk = blockIdx.x * 128;

  v8f acc[4][2];
#pragma unroll
  for (int mi = 0; mi < 4; ++mi)
#pragma unroll
    for (int ni = 0; ni < 2; ++ni) acc[mi][ni] = zero8();

  const int nk = K / GEMM_KSTAGE;
  stage_load(A, BT, K, m0blk, n0blk, 0, tileA[0], tileB[0]);
  if (nk > 1) stage_load(A, BT, K, m0blk, n0blk, 1, tileA[1], tileB[1]);

  for (int ks = 0; ks < nk; ++ks) {
    wait_async_stage(ks + 1 < nk);   // my stage-ks chunks have landed in LDS
    __syncthreads();                 // everyone's stage-ks chunks visible

    const __bf16 (*tA)[LDS_PAD] = tileA[ks & 1];
    const __bf16 (*tB)[LDS_PAD] = tileB[ks & 1];
#pragma unroll
    for (int kk = 0; kk < GEMM_KSTAGE; kk += 32) {
      v16bf a[4], b[2];
#pragma unroll
      for (int mi = 0; mi < 4; ++mi)
        a[mi] = load_a_frag(&tA[wm * 64 + mi * 16][0], LDS_PAD, kk);
#pragma unroll
      for (int ni = 0; ni < 2; ++ni)
        b[ni] = load_b_frag(&tB[wn * 32 + ni * 16][0], LDS_PAD, kk);
#pragma unroll
      for (int mi = 0; mi < 4; ++mi)
#pragma unroll
        for (int ni = 0; ni < 2; ++ni)
          acc[mi][ni] = wmma_bf16(a[mi], b[ni], acc[mi][ni]);
    }

    __syncthreads();                 // all waves done reading buffer ks&1
    if (ks + 2 < nk)
      stage_load(A, BT, K, m0blk, n0blk, ks + 2, tileA[ks & 1], tileB[ks & 1]);
  }

  const int m0 = m0blk + wm * 64;
  const int n0 = n0blk + wn * 32;
#pragma unroll
  for (int mi = 0; mi < 4; ++mi)
#pragma unroll
    for (int ni = 0; ni < 2; ++ni)
      store_c_frag(C + (long)(m0 + mi * 16) * N + n0 + ni * 16, N, acc[mi][ni]);
}

// ---------------------------------------------------------------------------
// Repack qkv / pe GEMM outputs into attention-friendly bf16 layouts:
//   Qu[q,c] = Q[q,c] + u[c],  Qv[q,c] = Q[q,c] + v[c]    (q = row-2048)
//   Kb[k,c] = K,  PEb[k,c] = pe,  Vt[c,k] = V[k,c]       (c = h*64+d)
// ---------------------------------------------------------------------------

__global__ void repack_kernel(const float* __restrict__ qkv, const float* __restrict__ pe,
                              const float* __restrict__ u, const float* __restrict__ v,
                              __bf16* __restrict__ Qu, __bf16* __restrict__ Qv,
                              __bf16* __restrict__ Kb, __bf16* __restrict__ PEb,
                              __bf16* __restrict__ Vt) {
  int i = blockIdx.x * blockDim.x + threadIdx.x;
  if (i >= K_LEN * D_MODEL) return;
  const int k = i >> 10;        // row 0..4095
  const int c = i & 1023;       // h*64+d
  const long qrow = (long)k * (3 * D_MODEL);
  Kb[i]  = (__bf16)qkv[qrow + D_MODEL + c];
  PEb[i] = (__bf16)pe[i];
  Vt[(long)c * K_LEN + k] = (__bf16)qkv[qrow + 2 * D_MODEL + c];
  if (k >= MEM_LEN) {
    const int q = k - MEM_LEN;
    const float qval = qkv[qrow + c];
    Qu[(long)q * D_MODEL + c] = (__bf16)(qval + u[c]);
    Qv[(long)q * D_MODEL + c] = (__bf16)(qval + v[c]);
  }
}

// ---------------------------------------------------------------------------
// Flash attention, one (head, 64-query tile) per 128-thread block (4 waves);
// each wave owns 16 query rows. Online softmax over 64-wide key tiles.
// Scores: S = (Q+u)K^T + (Q+v)PE^T, scaled by 1/8, causal mask j <= i+2048.
// K/PE/V fragments come straight from global (24 MB total, resident in the
// 192 MB L2 and reused by all 32 q-tiles per head).
// ---------------------------------------------------------------------------

__global__ __launch_bounds__(128) void attn_kernel(const __bf16* __restrict__ Qu,
                                                   const __bf16* __restrict__ Qv,
                                                   const __bf16* __restrict__ Kb,
                                                   const __bf16* __restrict__ PEb,
                                                   const __bf16* __restrict__ Vt,
                                                   __bf16* __restrict__ av) {
  __shared__ __align__(16) __bf16 pbuf[4][16][72];   // per-wave P tile, padded

  const int wave = threadIdx.x >> 5;
  const int lane = threadIdx.x & 31;
  const int g = lane >> 4, n16 = lane & 15;
  const int h = blockIdx.y;
  const int qb = blockIdx.x * 64;
  const int q0 = qb + wave * 16;

  // Persistent A-fragments: Qu/Qv rows q0..q0+15, d = 0..63 (two K-steps).
  const __bf16* quB = Qu + (long)q0 * D_MODEL + h * D_HEAD;
  const __bf16* qvB = Qv + (long)q0 * D_MODEL + h * D_HEAD;
  v16bf aqu0 = load_a_frag(quB, D_MODEL, 0), aqu1 = load_a_frag(quB, D_MODEL, 32);
  v16bf aqv0 = load_a_frag(qvB, D_MODEL, 0), aqv1 = load_a_frag(qvB, D_MODEL, 32);

  v8f o[4];
#pragma unroll
  for (int df = 0; df < 4; ++df) o[df] = zero8();
  float mrow[8], lrow[8];
#pragma unroll
  for (int r = 0; r < 8; ++r) { mrow[r] = -1e30f; lrow[r] = 0.0f; }

  const int ktiles = blockIdx.x + 33;   // last tile holds the causal diagonal
  for (int kt = 0; kt < ktiles; ++kt) {
    const int kb = kt * 64;

    // ---- scores: 4 N-fragments of 16x16 ----
    v8f s[4];
#pragma unroll
    for (int nf = 0; nf < 4; ++nf) {
      const __bf16* kbase = Kb  + (long)(kb + nf * 16) * D_MODEL + h * D_HEAD;
      const __bf16* pbase = PEb + (long)(kb + nf * 16) * D_MODEL + h * D_HEAD;
      v8f acc = zero8();
      acc = wmma_bf16(aqu0, load_b_frag(kbase, D_MODEL, 0),  acc);
      acc = wmma_bf16(aqu1, load_b_frag(kbase, D_MODEL, 32), acc);
      acc = wmma_bf16(aqv0, load_b_frag(pbase, D_MODEL, 0),  acc);
      acc = wmma_bf16(aqv1, load_b_frag(pbase, D_MODEL, 32), acc);
      s[nf] = acc;
    }

    // ---- scale + causal mask (j > i + MEM_LEN -> masked) ----
#pragma unroll
    for (int nf = 0; nf < 4; ++nf) {
#pragma unroll
      for (int r = 0; r < 8; ++r) {
        const int qi = q0 + r + 8 * g;
        const int j  = kb + nf * 16 + n16;
        float sv = s[nf][r] * 0.125f;
        s[nf][r] = (j > qi + MEM_LEN) ? -1e30f : sv;
      }
    }

    // ---- online softmax (rows live across the 16-lane half-wave) ----
    float corr[8];
#pragma unroll
    for (int r = 0; r < 8; ++r) {
      float t = fmaxf(fmaxf(s[0][r], s[1][r]), fmaxf(s[2][r], s[3][r]));
#pragma unroll
      for (int off = 1; off < 16; off <<= 1) t = fmaxf(t, __shfl_xor(t, off, 16));
      const float mn = fmaxf(mrow[r], t);
      corr[r] = __expf(mrow[r] - mn);
      mrow[r] = mn;
      float acc = 0.0f;
#pragma unroll
      for (int nf = 0; nf < 4; ++nf) {
        float p = __expf(s[nf][r] - mn);
        s[nf][r] = p;
        acc += p;
      }
#pragma unroll
      for (int off = 1; off < 16; off <<= 1) acc += __shfl_xor(acc, off, 16);
      lrow[r] = lrow[r] * corr[r] + acc;
    }
#pragma unroll
    for (int df = 0; df < 4; ++df)
#pragma unroll
      for (int r = 0; r < 8; ++r) o[df][r] *= corr[r];

    // ---- P -> bf16 A-fragment via per-wave LDS (wave-internal, in-order) ----
#pragma unroll
    for (int nf = 0; nf < 4; ++nf)
#pragma unroll
      for (int r = 0; r < 8; ++r)
        pbuf[wave][r + 8 * g][nf * 16 + n16] = (__bf16)s[nf][r];
    v16bf pa0 = load_a_frag(&pbuf[wave][0][0], 72, 0);
    v16bf pa1 = load_a_frag(&pbuf[wave][0][0], 72, 32);

    // ---- O += P * V  (V^T layout: Vt[d][k], contiguous k per lane) ----
#pragma unroll
    for (int df = 0; df < 4; ++df) {
      const __bf16* vbase = Vt + (long)(h * D_HEAD + df * 16) * K_LEN + kb;
      o[df] = wmma_bf16(pa0, load_b_frag(vbase, K_LEN, 0),  o[df]);
      o[df] = wmma_bf16(pa1, load_b_frag(vbase, K_LEN, 32), o[df]);
    }
  }

  // ---- normalize and emit bf16 av tile ----
#pragma unroll
  for (int df = 0; df < 4; ++df)
#pragma unroll
    for (int r = 0; r < 8; ++r) {
      const int qi = q0 + r + 8 * g;
      av[(long)qi * D_MODEL + h * D_HEAD + df * 16 + n16] = (__bf16)(o[df][r] / lrow[r]);
    }
}

// ---------------------------------------------------------------------------
// Residual + LayerNorm: out = LN(hidden + attn), eps = 1e-5, no affine.
// One 256-thread block per row.
// ---------------------------------------------------------------------------

__global__ __launch_bounds__(256) void ln_kernel(const float* __restrict__ hidden,
                                                 const float* __restrict__ attn,
                                                 float* __restrict__ out) {
  __shared__ float red[2][8];
  const int q = blockIdx.x;
  const int wave = threadIdx.x >> 5, lane = threadIdx.x & 31;
  float s = 0.0f, ss = 0.0f;
  for (int c = threadIdx.x; c < D_MODEL; c += 256) {
    const float x = hidden[(long)q * D_MODEL + c] + attn[(long)q * D_MODEL + c];
    s += x; ss += x * x;
  }
#pragma unroll
  for (int off = 16; off >= 1; off >>= 1) {
    s  += __shfl_xor(s,  off, 32);
    ss += __shfl_xor(ss, off, 32);
  }
  if (lane == 0) { red[0][wave] = s; red[1][wave] = ss; }
  __syncthreads();
  if (threadIdx.x == 0) {
    float a = 0.0f, b = 0.0f;
#pragma unroll
    for (int i = 0; i < 8; ++i) { a += red[0][i]; b += red[1][i]; }
    red[0][0] = a; red[1][0] = b;
  }
  __syncthreads();
  const float mu  = red[0][0] * (1.0f / D_MODEL);
  const float var = red[1][0] * (1.0f / D_MODEL) - mu * mu;
  const float rs  = rsqrtf(var + 1e-5f);
  for (int c = threadIdx.x; c < D_MODEL; c += 256) {
    const float x = hidden[(long)q * D_MODEL + c] + attn[(long)q * D_MODEL + c];
    out[(long)q * D_MODEL + c] = (x - mu) * rs;
  }
}

// ---------------------------------------------------------------------------
// Host orchestration
// ---------------------------------------------------------------------------

extern "C" void kernel_launch(void* const* d_in, const int* in_sizes, int n_in,
                              void* d_out, int out_size, void* d_ws, size_t ws_size,
                              hipStream_t stream) {
  const float* hidden = (const float*)d_in[0];   // (1, 2048, 1024)
  const float* pos    = (const float*)d_in[1];   // (4096, 1024)
  const float* u      = (const float*)d_in[2];   // (16, 64)
  const float* v      = (const float*)d_in[3];   // (16, 64)
  // d_in[4] = mask: causal structure is analytic (j > i + MEM_LEN), ignored.
  const float* memory = (const float*)d_in[5];   // (1, 2048, 1024)
  const float* W_qkv  = (const float*)d_in[6];   // (1024, 3072)
  const float* W_pe   = (const float*)d_in[7];   // (1024, 1024)
  const float* W_out  = (const float*)d_in[8];   // (1024, 1024)
  float* out = (float*)d_out;

  // ---- carve workspace (≈134 MB) ----
  char* ws = (char*)d_ws;
  size_t off = 0;
  auto carve = [&](size_t bytes) -> void* {
    void* p = ws + off;
    off = (off + bytes + 255) & ~(size_t)255;
    return p;
  };
  __bf16* mhB   = (__bf16*)carve((size_t)K_LEN * D_MODEL * 2);        // concat(memory, hidden)
  __bf16* peInB = (__bf16*)carve((size_t)K_LEN * D_MODEL * 2);
  __bf16* WqkvT = (__bf16*)carve((size_t)3 * D_MODEL * D_MODEL * 2);  // (3072,1024)
  __bf16* WpeT  = (__bf16*)carve((size_t)D_MODEL * D_MODEL * 2);
  __bf16* WoutT = (__bf16*)carve((size_t)D_MODEL * D_MODEL * 2);
  float*  qkvF  = (float*)carve((size_t)K_LEN * 3 * D_MODEL * 4);
  float*  peF   = (float*)carve((size_t)K_LEN * D_MODEL * 4);
  __bf16* QuB   = (__bf16*)carve((size_t)Q_LEN * D_MODEL * 2);
  __bf16* QvB   = (__bf16*)carve((size_t)Q_LEN * D_MODEL * 2);
  __bf16* KB    = (__bf16*)carve((size_t)K_LEN * D_MODEL * 2);
  __bf16* PEB   = (__bf16*)carve((size_t)K_LEN * D_MODEL * 2);
  __bf16* VtB   = (__bf16*)carve((size_t)D_MODEL * K_LEN * 2);        // (h*64+d, k)
  __bf16* avB   = (__bf16*)carve((size_t)Q_LEN * D_MODEL * 2);
  float*  attnF = (float*)carve((size_t)Q_LEN * D_MODEL * 4);
  if (off > ws_size) return;   // workspace too small: bail deterministically

  const int nElemHalf = MEM_LEN * D_MODEL;   // 2M
  const int nElemFull = K_LEN * D_MODEL;     // 4M

  // 1. fp32 -> bf16 converts (mh = [memory; hidden]) and weight transposes.
  cvt_f32_bf16<<<(nElemHalf + 255) / 256, 256, 0, stream>>>(memory, mhB, nElemHalf);
  cvt_f32_bf16<<<(nElemHalf + 255) / 256, 256, 0, stream>>>(hidden, mhB + nElemHalf, nElemHalf);
  cvt_f32_bf16<<<(nElemFull + 255) / 256, 256, 0, stream>>>(pos, peInB, nElemFull);
  transpose_cvt<<<dim3(3 * D_MODEL / 32, D_MODEL / 32), dim3(32, 8), 0, stream>>>(
      W_qkv, WqkvT, D_MODEL, 3 * D_MODEL);
  transpose_cvt<<<dim3(D_MODEL / 32, D_MODEL / 32), dim3(32, 8), 0, stream>>>(
      W_pe, WpeT, D_MODEL, D_MODEL);
  transpose_cvt<<<dim3(D_MODEL / 32, D_MODEL / 32), dim3(32, 8), 0, stream>>>(
      W_out, WoutT, D_MODEL, D_MODEL);

  // 2. qkv = mh @ W_qkv  (4096 x 3072 x 1024),  pe = pos @ W_pe (4096 x 1024 x 1024)
  gemm_bf16<<<dim3(3 * D_MODEL / 128, K_LEN / 128), 256, 0, stream>>>(
      mhB, WqkvT, qkvF, K_LEN, 3 * D_MODEL, D_MODEL);
  gemm_bf16<<<dim3(D_MODEL / 128, K_LEN / 128), 256, 0, stream>>>(
      peInB, WpeT, peF, K_LEN, D_MODEL, D_MODEL);

  // 3. repack into Qu/Qv/K/PE/V^T bf16 layouts
  repack_kernel<<<(nElemFull + 255) / 256, 256, 0, stream>>>(
      qkvF, peF, u, v, QuB, QvB, KB, PEB, VtB);

  // 4. flash attention: grid = (q-tiles, heads)
  attn_kernel<<<dim3(Q_LEN / 64, N_HEAD), 128, 0, stream>>>(QuB, QvB, KB, PEB, VtB, avB);

  // 5. attn_out = av @ W_out  (2048 x 1024 x 1024)
  gemm_bf16<<<dim3(D_MODEL / 128, Q_LEN / 128), 256, 0, stream>>>(
      avB, WoutT, attnF, Q_LEN, D_MODEL, D_MODEL);

  // 6. residual + LayerNorm
  ln_kernel<<<Q_LEN, 256, 0, stream>>>(hidden, attnF, out);
}
